// Model_39676907883315
// MI455X (gfx1250) — matrix operand order repaired and checked by reading, not compiled, not run on hardware
//
#include <hip/hip_runtime.h>
#include <hip/hip_bf16.h>

typedef __attribute__((ext_vector_type(16))) __bf16 v16bf;
typedef __attribute__((ext_vector_type(8)))  float  v8f;

#define B_   8
#define SQ_  2048
#define SKV_ 2048
#define D_   1024

constexpr float kInvScale = 1.0f / 0.72f;

// round-to-nearest-even f32 -> bf16
__device__ __forceinline__ unsigned short f2bf(float f) {
  unsigned u = __float_as_uint(f);
  u += 0x7FFFu + ((u >> 16) & 1u);
  return (unsigned short)(u >> 16);
}

union FragBf {
  v16bf v;
  unsigned short u[16];
  uint4 q[2];          // q[0] = elements 0..7, q[1] = elements 8..15
};

// A-frag (16x32 bf16): lane half h covers K = h*16..h*16+15 (elems 0..15)
// B-frag (32x16 bf16): lane half h covers K = h*16..h*16+15 (elems 0..15)
// -> with K contiguous in memory, each frag is exactly two 16-byte loads.

__device__ __forceinline__ v8f wmma_bf16(const FragBf& a, const FragBf& b, v8f c) {
  return __builtin_amdgcn_wmma_f32_16x16x32_bf16(false, a.v, false, b.v,
                                                 (short)0, c, false, false);
}

// ---------------------------------------------------------------------------
// Pre-pass 1: streaming f32 -> bf16 convert (for Q and K)
// ---------------------------------------------------------------------------
__global__ void __launch_bounds__(256)
cvt_bf16(const float* __restrict__ src, unsigned int* __restrict__ dst) {
  const int i = blockIdx.x * 256 + threadIdx.x;        // index in f32 pairs
  float2 f = ((const float2*)src)[i];
  dst[i] = (unsigned)f2bf(f.x) | ((unsigned)f2bf(f.y) << 16);
}

// ---------------------------------------------------------------------------
// Pre-pass 2: tiled transpose f32[rows][cols] -> bf16[cols][rows] (V and W)
// ---------------------------------------------------------------------------
__global__ void __launch_bounds__(256)
transpose_bf16(const float* __restrict__ src, unsigned short* __restrict__ dst,
               int rows, int cols) {
  __shared__ unsigned short tile[32][33];
  const size_t base = (size_t)blockIdx.z * rows * cols;
  const int c0 = blockIdx.x * 32, r0 = blockIdx.y * 32;
  const int tx = threadIdx.x, ty = threadIdx.y;        // 32 x 8
#pragma unroll
  for (int i = ty; i < 32; i += 8)
    tile[i][tx] = f2bf(src[base + (size_t)(r0 + i) * cols + c0 + tx]);
  __syncthreads();
#pragma unroll
  for (int i = ty; i < 32; i += 8)
    dst[base + (size_t)(c0 + i) * rows + r0 + tx] = tile[tx][i];
}

// ---------------------------------------------------------------------------
// Kernel 1: fused attention. Block = 256 threads (8 waves), 16 query rows.
//   LDS: sS = f32 scores [16][2048] (128 KB), sP = bf16 probs (64 KB)
// ---------------------------------------------------------------------------
__global__ void __launch_bounds__(256)
attn_kernel(const unsigned short* __restrict__ qBf,
            const unsigned short* __restrict__ kBf,
            const unsigned short* __restrict__ vtBf,   // [B][D][SKV] bf16
            const float* __restrict__ drop,
            const float* __restrict__ bias,
            unsigned short* __restrict__ oBf) {
  extern __shared__ float sS[];                        // 16 * SKV_ f32
  unsigned short* sP = (unsigned short*)(sS + 16 * SKV_);
  const int b    = blockIdx.y;
  const int q0   = blockIdx.x * 16;
  const int tid  = threadIdx.x;
  const int w    = tid >> 5;
  const int lane = tid & 31;
  const int m    = lane & 15;
  const int h    = lane >> 4;

  // ---- Phase 1: S = Q Kt / 0.72 -------------------------------------------
  {
    const unsigned short* qRow = qBf + ((size_t)b * SQ_ + q0 + m) * D_;
    for (int it = 0; it < 8; ++it) {                   // 2 kv tiles / iter
      const int kv0 = w * 256 + it * 32;
      const unsigned short* kRow0 = kBf + ((size_t)b * SKV_ + kv0 + m) * D_;
      const unsigned short* kRow1 = kRow0 + 16 * D_;
      v8f acc0 = {}, acc1 = {};
      for (int dc = 0; dc < 32; ++dc) {
        const int d0 = dc * 32;
        FragBf fa, fb0, fb1;
        fa.q[0]  = *(const uint4*)(qRow + d0 + h * 16);
        fa.q[1]  = *(const uint4*)(qRow + d0 + h * 16 + 8);
        fb0.q[0] = *(const uint4*)(kRow0 + d0 + h * 16);
        fb0.q[1] = *(const uint4*)(kRow0 + d0 + h * 16 + 8);
        fb1.q[0] = *(const uint4*)(kRow1 + d0 + h * 16);
        fb1.q[1] = *(const uint4*)(kRow1 + d0 + h * 16 + 8);
        acc0 = wmma_bf16(fa, fb0, acc0);
        acc1 = wmma_bf16(fa, fb1, acc1);
      }
#pragma unroll
      for (int j = 0; j < 8; ++j) {
        sS[(j + 8 * h) * SKV_ + kv0 + m]      = acc0[j] * kInvScale;
        sS[(j + 8 * h) * SKV_ + kv0 + 16 + m] = acc1[j] * kInvScale;
      }
    }
  }
  __syncthreads();

  // ---- Phase 2: softmax, then * drop_mask, pack bf16 to sP ----------------
  {
    const int r = tid >> 4;                            // 16 rows, 16 thr/row
    const int s = tid & 15;
    float mx = -3.0e38f;
    for (int c = s; c < SKV_; c += 16) mx = fmaxf(mx, sS[r * SKV_ + c]);
    mx = fmaxf(mx, __shfl_xor(mx, 1));
    mx = fmaxf(mx, __shfl_xor(mx, 2));
    mx = fmaxf(mx, __shfl_xor(mx, 4));
    mx = fmaxf(mx, __shfl_xor(mx, 8));
    float sum = 0.0f;
    for (int c = s; c < SKV_; c += 16) {
      float e = __expf(sS[r * SKV_ + c] - mx);
      sS[r * SKV_ + c] = e;
      sum += e;
    }
    sum += __shfl_xor(sum, 1);
    sum += __shfl_xor(sum, 2);
    sum += __shfl_xor(sum, 4);
    sum += __shfl_xor(sum, 8);
    const float inv = 1.0f / sum;
    const float* dRow = drop + ((size_t)b * SQ_ + q0 + r) * SKV_;
    for (int c = s; c < SKV_; c += 16)
      sP[r * SKV_ + c] = f2bf(sS[r * SKV_ + c] * inv * dRow[c]);
  }
  __syncthreads();

  // ---- Phase 3: O = P V + bias --------------------------------------------
  {
    const unsigned short* pRow = sP + m * SKV_;
    for (int it = 0; it < 4; ++it) {                   // 2 col tiles / iter
      const int col0 = w * 128 + it * 32;
      const unsigned short* vtRow0 = vtBf + ((size_t)b * D_ + col0 + m) * SKV_;
      const unsigned short* vtRow1 = vtRow0 + 16 * SKV_;
      v8f acc0 = {}, acc1 = {};
      for (int kc = 0; kc < 64; ++kc) {
        const int k0 = kc * 32;
        FragBf fa, fb0, fb1;
        fa.q[0]  = *(const uint4*)(pRow + k0 + h * 16);      // ds_load_b128
        fa.q[1]  = *(const uint4*)(pRow + k0 + h * 16 + 8);
        fb0.q[0] = *(const uint4*)(vtRow0 + k0 + h * 16);
        fb0.q[1] = *(const uint4*)(vtRow0 + k0 + h * 16 + 8);
        fb1.q[0] = *(const uint4*)(vtRow1 + k0 + h * 16);
        fb1.q[1] = *(const uint4*)(vtRow1 + k0 + h * 16 + 8);
        acc0 = wmma_bf16(fa, fb0, acc0);
        acc1 = wmma_bf16(fa, fb1, acc1);
      }
      const float bias0 = bias[col0 + m];
      const float bias1 = bias[col0 + 16 + m];
#pragma unroll
      for (int j = 0; j < 8; ++j) {
        const int rr = j + 8 * h;
        unsigned short* oRow = oBf + ((size_t)b * SQ_ + q0 + rr) * D_;
        oRow[col0 + m]      = f2bf(acc0[j] + bias0);
        oRow[col0 + 16 + m] = f2bf(acc1[j] + bias1);
      }
    }
  }
}

// ---------------------------------------------------------------------------
// Kernel 2: out[16384][1024] = oBf @ W  (B-frags from wtBf = W^T bf16).
// 8 waves/block; each wave computes a 32x32 tile (2x2 WMMA blocking).
// ---------------------------------------------------------------------------
__global__ void __launch_bounds__(256)
proj_kernel(const unsigned short* __restrict__ oBf,
            const unsigned short* __restrict__ wtBf,   // [E][D] bf16 (= W^T)
            float* __restrict__ out) {
  const int tid  = threadIdx.x;
  const int w    = tid >> 5;
  const int lane = tid & 31;
  const int m    = lane & 15;
  const int h    = lane >> 4;
  const int blk  = blockIdx.x * 8 + w;
  const int mb   = blk >> 5;                           // 32 col-blocks of 32
  const int nb   = blk & 31;
  const int row0 = mb * 32;
  const int col0 = nb * 32;

  const unsigned short* aRow0 = oBf  + (size_t)(row0 + m) * D_;
  const unsigned short* aRow1 = aRow0 + (size_t)16 * D_;
  const unsigned short* bRow0 = wtBf + (size_t)(col0 + m) * D_;
  const unsigned short* bRow1 = bRow0 + (size_t)16 * D_;

  v8f acc00 = {}, acc01 = {}, acc10 = {}, acc11 = {};
  for (int kc = 0; kc < 32; ++kc) {
    const int k0 = kc * 32;
    FragBf fa0, fa1, fb0, fb1;
    fa0.q[0] = *(const uint4*)(aRow0 + k0 + h * 16);
    fa0.q[1] = *(const uint4*)(aRow0 + k0 + h * 16 + 8);
    fa1.q[0] = *(const uint4*)(aRow1 + k0 + h * 16);
    fa1.q[1] = *(const uint4*)(aRow1 + k0 + h * 16 + 8);
    fb0.q[0] = *(const uint4*)(bRow0 + k0 + h * 16);
    fb0.q[1] = *(const uint4*)(bRow0 + k0 + h * 16 + 8);
    fb1.q[0] = *(const uint4*)(bRow1 + k0 + h * 16);
    fb1.q[1] = *(const uint4*)(bRow1 + k0 + h * 16 + 8);
    acc00 = wmma_bf16(fa0, fb0, acc00);
    acc01 = wmma_bf16(fa0, fb1, acc01);
    acc10 = wmma_bf16(fa1, fb0, acc10);
    acc11 = wmma_bf16(fa1, fb1, acc11);
  }
#pragma unroll
  for (int j = 0; j < 8; ++j) {
    const int rr = j + 8 * h;
    float* oA = out + (size_t)(row0 + rr) * D_;
    float* oB = out + (size_t)(row0 + 16 + rr) * D_;
    oA[col0 + m]      = acc00[j];
    oA[col0 + 16 + m] = acc01[j];
    oB[col0 + m]      = acc10[j];
    oB[col0 + 16 + m] = acc11[j];
  }
}

// ---------------------------------------------------------------------------
extern "C" void kernel_launch(void* const* d_in, const int* in_sizes, int n_in,
                              void* d_out, int out_size, void* d_ws, size_t ws_size,
                              hipStream_t stream) {
  const float* q1      = (const float*)d_in[0];
  const float* key     = (const float*)d_in[1];
  const float* value   = (const float*)d_in[2];
  const float* weights = (const float*)d_in[3];
  const float* bias    = (const float*)d_in[4];
  const float* drop    = (const float*)d_in[5];
  float* out = (float*)d_out;

  constexpr size_t nQKV = (size_t)B_ * SQ_ * D_;       // 16.78M elements
  char* ws = (char*)d_ws;
  unsigned short* oBf  = (unsigned short*)(ws);                       // 32 MB
  unsigned short* qBf  = (unsigned short*)(ws + nQKV * 2);            // 32 MB
  unsigned short* kBf  = (unsigned short*)(ws + nQKV * 4);            // 32 MB
  unsigned short* vtBf = (unsigned short*)(ws + nQKV * 6);            // 32 MB
  unsigned short* wtBf = (unsigned short*)(ws + nQKV * 8);            //  2 MB

  // pre-passes: convert Q,K; transpose-convert V,W
  cvt_bf16<<<(int)(nQKV / 2 / 256), 256, 0, stream>>>(q1,  (unsigned int*)qBf);
  cvt_bf16<<<(int)(nQKV / 2 / 256), 256, 0, stream>>>(key, (unsigned int*)kBf);
  transpose_bf16<<<dim3(D_ / 32, SKV_ / 32, B_), dim3(32, 8), 0, stream>>>(
      value, vtBf, SKV_, D_);
  transpose_bf16<<<dim3(D_ / 32, D_ / 32, 1), dim3(32, 8), 0, stream>>>(
      weights, wtBf, D_, D_);

  // fused attention (192 KB dynamic LDS: 128 KB f32 scores + 64 KB bf16 probs)
  attn_kernel<<<dim3(SQ_ / 16, B_), 256,
                16 * SKV_ * (int)sizeof(float) + 16 * SKV_ * (int)sizeof(short),
                stream>>>(qBf, kBf, vtBf, drop, bias, oBf);

  // projection GEMM
  proj_kernel<<<((B_ * SQ_ / 32) * (D_ / 32)) / 8, 256, 0, stream>>>(oBf, wtBf, out);
}
